// TrajGRU_31997506355909
// MI455X (gfx1250) — matrix operand order repaired and checked by reading, compiled, not run on hardware
//
#include <hip/hip_runtime.h>
#include <hip/hip_bf16.h>

// ---------------------------------------------------------------------------
// TrajGRU on MI455X (gfx1250): implicit-GEMM convs on v_wmma_f32_16x16x32_bf16
// NHWC bf16 activations + (tap,ci)-ordered K so B fragments load as v16bf.
// M-tiled: each wave holds MT accumulators, reusing one B fragment.
// ---------------------------------------------------------------------------

typedef __attribute__((ext_vector_type(16))) __bf16 v16bf;
typedef __attribute__((ext_vector_type(8)))  float  v8f;

static constexpr int Bn   = 4;     // batch
static constexpr int CIN0 = 8;     // input channels
static constexpr int Fch  = 96;    // hidden channels
static constexpr int Hdim = 96;
static constexpr int Wdim = 96;
static constexpr int Lfl  = 5;     // flow links
static constexpr int HW   = Hdim * Wdim;        // 9216
static constexpr int BFHW = Bn * Fch * HW;      // 3,538,944

// ---- WMMA fragment index maps (CDNA5 ISA §7.12.2, wave32) -------
// A (16-bit, 16x32 MxK): lane L holds M=L%16; this code uses K = 16*(L/16) + h (matches B).
__device__ __forceinline__ int a_frag_k(int lane, int h) {
    int grp = lane >> 4;          // 0: K 0..15, 1: K 16..31
    return 16 * grp + h;
}
// B (16-bit, 32x16 KxN): lane L holds N=L%16; this code uses K = 16*(L/16) + h.
// C/D (f32 16x16):       lane L holds N=L%16, M = vgpr + 8*(L/16).

__device__ __forceinline__ int iclamp(int v, int lo, int hi) {
    return v < lo ? lo : (v > hi ? hi : v);
}

// ---------------------------------------------------------------------------
// Implicit-GEMM conv, NHWC activations, weights repacked [COUT][KH*KW*CIN]
// (K index = tap*CIN + ci). Block: 256 threads = 8 waves;
// (16*MT) output channels x 128 pixels per block.
// grid = (HW/128, ceil(COUT/(16*MT)), N_images). Output is NCHW f32.
// ---------------------------------------------------------------------------
template <int CIN, int KH, int KW, int PAD, int MT>
__global__ __launch_bounds__(256) void conv_wmma(
    const __bf16* __restrict__ xin,   // [N][H][W][CIN] bf16 (NHWC)
    const __bf16* __restrict__ wgt,   // [COUT][KH*KW*CIN] bf16
    const float*  __restrict__ bias,  // [COUT]
    float*        __restrict__ yout,  // [N][COUT][H][W] f32 (NCHW)
    int COUT, int addTo, int leaky)
{
    constexpr int K    = CIN * KH * KW;
    constexpr int KC   = (K + 31) / 32;
    constexpr bool VEC = (CIN % 32) == 0;   // lane's 16 B values contiguous

    __shared__ __attribute__((aligned(32))) __bf16 ldsA[MT * 512];

    const int tid  = threadIdx.x;
    const int lane = tid & 31;
    const int wid  = tid >> 5;
    const int grp  = lane >> 4;

    const int pixBase = blockIdx.x * 128 + wid * 16;
    const int chBase  = blockIdx.y * (16 * MT);
    const int img     = blockIdx.z;

    const int ncol = lane & 15;              // B/C column
    const int pix  = pixBase + ncol;
    const int py   = pix / Wdim;
    const int px   = pix - py * Wdim;

    v8f acc[MT];
#pragma unroll
    for (int mt = 0; mt < MT; ++mt) acc[mt] = (v8f){};

    for (int kc = 0; kc < KC; ++kc) {
        // ---- stage MT A subtiles into LDS in *fragment order* --------------
#pragma unroll
        for (int e = 0; e < 2 * MT; ++e) {
            int idx = tid * (2 * MT) + e;    // [0, MT*512)
            int mt  = idx >> 9;
            int w9  = idx & 511;
            int sl  = w9 >> 4;               // fragment lane
            int sh  = w9 & 15;               // fragment half
            int m   = sl & 15;
            int kg  = kc * 32 + a_frag_k(sl, sh);
            int ch  = chBase + mt * 16 + m;
            bool ok = (kg < K) && (ch < COUT);
            int  kgc = kg < K ? kg : (K - 1);
            int  chc = ch < COUT ? ch : (COUT - 1);
            __bf16 v = wgt[(size_t)chc * K + kgc];
            ldsA[idx] = ok ? v : (__bf16)0.0f;
        }
        __syncthreads();

        // gfx1250 prefetch of next weight K-chunk (global_prefetch_b8)
        if ((kc + 1) < KC && lane == 0)
            __builtin_prefetch(&wgt[(size_t)chBase * K + (kc + 1) * 32], 0, 1);

        // ---- B fragment: per-lane im2col, reused MT times ------------------
        v16bf bfrag;
        if constexpr (VEC) {
            // K%32==0, lane's 16 values share one tap, contiguous ci.
            int kgbase = kc * 32 + (grp << 4);
            int tap    = kgbase / CIN;       // constant-folded division
            int ci0    = kgbase - tap * CIN; // multiple of 16 -> 32B aligned
            int kh     = tap / KW;
            int kw     = tap - kh * KW;
            int iy     = py + kh - PAD;
            int ix     = px + kw - PAD;
            bool ok = (iy >= 0) && (iy < Hdim) && (ix >= 0) && (ix < Wdim);
            int iyc = iclamp(iy, 0, Hdim - 1);
            int ixc = iclamp(ix, 0, Wdim - 1);
            const v16bf* src = (const v16bf*)
                &xin[(((size_t)img * HW) + iyc * Wdim + ixc) * CIN + ci0];
            v16bf ldv = *src;                // 32B: global_load_b128 x2
            v16bf zv = {};
            bfrag = ok ? ldv : zv;
        } else {
#pragma unroll
            for (int hh = 0; hh < 16; ++hh) {
                int kg  = kc * 32 + (grp << 4) + hh;
                int kgc = kg < K ? kg : (K - 1);
                int tap = kgc / CIN;
                int ci  = kgc - tap * CIN;
                int kh  = tap / KW;
                int kw  = tap - kh * KW;
                int iy  = py + kh - PAD;
                int ix  = px + kw - PAD;
                bool ok = (kg < K) && (iy >= 0) && (iy < Hdim) &&
                          (ix >= 0) && (ix < Wdim);
                int iyc = iclamp(iy, 0, Hdim - 1);
                int ixc = iclamp(ix, 0, Wdim - 1);
                __bf16 v =
                    xin[(((size_t)img * HW) + iyc * Wdim + ixc) * CIN + ci];
                bfrag[hh] = ok ? v : (__bf16)0.0f;
            }
        }

#pragma unroll
        for (int mt = 0; mt < MT; ++mt) {
            v16bf afrag = *(const v16bf*)&ldsA[mt * 512 + lane * 16];
            acc[mt] = __builtin_amdgcn_wmma_f32_16x16x32_bf16(
                false, afrag, false, bfrag, (short)0, acc[mt], false, false);
        }
        __syncthreads();
    }

    // ---- epilogue: bias (+optional accumulate, optional leaky-ReLU) --------
#pragma unroll
    for (int mt = 0; mt < MT; ++mt) {
#pragma unroll
        for (int r = 0; r < 8; ++r) {
            int ch = chBase + mt * 16 + r + 8 * grp;
            if (ch < COUT) {
                float v = acc[mt][r] + bias[ch];
                size_t o = ((size_t)img * COUT + ch) * HW + pix;
                if (addTo) v += yout[o];
                if (leaky) v = v > 0.0f ? v : 0.2f * v;
                yout[o] = v;
            }
        }
    }
}

// ---------------------------------------------------------------------------
// NCHW f32 -> NHWC bf16 transpose-convert. idx maps to output (c fastest)
// so stores are fully coalesced.
// ---------------------------------------------------------------------------
__global__ void nchw_to_nhwc_bf16(const float* __restrict__ src,  // [N][C][HW]
                                  __bf16* __restrict__ dst,       // [N][HW][C]
                                  int C, int total)               // total=N*C*HW
{
    int idx = blockIdx.x * blockDim.x + threadIdx.x;
    if (idx >= total) return;
    int c = idx % C;
    int t = idx / C;
    int p = t % HW;
    int n = t / HW;
    dst[idx] = (__bf16)src[((size_t)n * C + c) * HW + p];
}

__global__ void cvt_f32_bf16(const float* __restrict__ src,
                             __bf16* __restrict__ dst, int n)
{
    int i = blockIdx.x * blockDim.x + threadIdx.x;
    if (i < n) dst[i] = (__bf16)src[i];
}

// Repack OIHW f32 weights -> [COUT][KH][KW][CIN] bf16 (K = tap*CIN + ci).
__global__ void repack_w_bf16(const float* __restrict__ src,  // [O][I][KH][KW]
                              __bf16* __restrict__ dst,       // [O][KH*KW][I]
                              int CI, int KHW, int total)     // total=O*KHW*CI
{
    int idx = blockIdx.x * blockDim.x + threadIdx.x;
    if (idx >= total) return;
    int ci  = idx % CI;
    int t   = idx / CI;
    int tap = t % KHW;
    int co  = t / KHW;
    dst[idx] = (__bf16)src[((size_t)co * CI + ci) * KHW + tap];
}

__global__ void copy_f32(const float* __restrict__ src,
                         float* __restrict__ dst, int n)
{
    int i = blockIdx.x * blockDim.x + threadIdx.x;
    if (i < n) dst[i] = src[i];
}

// ---------------------------------------------------------------------------
// Flow warp: NHWC output. warped[b][p][l*F+c] = bilinear(h_prev[b][c], grid).
// torch grid_sample semantics: zeros padding, align_corners=False.
// ---------------------------------------------------------------------------
__global__ __launch_bounds__(256) void warp_kernel(
    const float* __restrict__ h_prev,   // [B][F][H][W] f32 (NCHW)
    const float* __restrict__ flows,    // [B][2L][H][W] f32 (NCHW)
    __bf16*      __restrict__ warped)   // [B][HW][L*F] bf16 (NHWC)
{
    int idx = blockIdx.x * blockDim.x + threadIdx.x;
    if (idx >= Bn * Lfl * HW) return;
    int p = idx % HW;
    int t = idx / HW;
    int l = t % Lfl;
    int b = t / Lfl;
    int y = p / Wdim, x = p - y * Wdim;

    float fx = -flows[(((size_t)b * (2 * Lfl)) + 2 * l + 0) * HW + p];
    float fy = -flows[(((size_t)b * (2 * Lfl)) + 2 * l + 1) * HW + p];
    float vx = (float)x + fx;
    float vy = (float)y + fy;
    float gx = 2.0f * vx / (float)(Wdim - 1) - 1.0f;
    float gy = 2.0f * vy / (float)(Hdim - 1) - 1.0f;
    float ix = ((gx + 1.0f) * (float)Wdim - 1.0f) * 0.5f;
    float iy = ((gy + 1.0f) * (float)Hdim - 1.0f) * 0.5f;

    int ix0 = (int)floorf(ix), iy0 = (int)floorf(iy);
    int ix1 = ix0 + 1,          iy1 = iy0 + 1;
    float wx1 = ix - (float)ix0, wy1 = iy - (float)iy0;
    float wx0 = 1.0f - wx1,      wy0 = 1.0f - wy1;

    bool vx0 = (ix0 >= 0) && (ix0 < Wdim);
    bool vx1 = (ix1 >= 0) && (ix1 < Wdim);
    bool vy0 = (iy0 >= 0) && (iy0 < Hdim);
    bool vy1 = (iy1 >= 0) && (iy1 < Hdim);
    float w00 = (vy0 && vx0) ? wy0 * wx0 : 0.0f;
    float w01 = (vy0 && vx1) ? wy0 * wx1 : 0.0f;
    float w10 = (vy1 && vx0) ? wy1 * wx0 : 0.0f;
    float w11 = (vy1 && vx1) ? wy1 * wx1 : 0.0f;
    int xc0 = iclamp(ix0, 0, Wdim - 1), xc1 = iclamp(ix1, 0, Wdim - 1);
    int yc0 = iclamp(iy0, 0, Hdim - 1), yc1 = iclamp(iy1, 0, Hdim - 1);
    int o00 = yc0 * Wdim + xc0, o01 = yc0 * Wdim + xc1;
    int o10 = yc1 * Wdim + xc0, o11 = yc1 * Wdim + xc1;

    const float* src = h_prev + (size_t)b * Fch * HW;
    __bf16* dst = warped + ((size_t)b * HW + p) * (Lfl * Fch) + l * Fch;
    for (int c = 0; c < Fch; ++c) {
        const float* ch = src + (size_t)c * HW;
        float a = w00 * ch[o00] + w01 * ch[o01] +
                  w10 * ch[o10] + w11 * ch[o11];
        dst[c] = (__bf16)a;   // contiguous NHWC channel run
    }
}

// ---------------------------------------------------------------------------
// GRU gate fusion: next_h = u*prev + (1-u)*leaky(m + r*h_m)
// ---------------------------------------------------------------------------
__device__ __forceinline__ float sigmoidf(float x) {
    return 1.0f / (1.0f + __expf(-x));
}

__global__ __launch_bounds__(256) void gate_kernel(
    const float* __restrict__ i2h,    // [B][3F][HW]
    const float* __restrict__ h2h,    // [B][3F][HW]
    const float* __restrict__ hprev,  // [B][F][HW]
    float*       __restrict__ outT)   // [B][F][HW]
{
    int idx = blockIdx.x * blockDim.x + threadIdx.x;
    if (idx >= BFHW) return;
    int p = idx % HW;
    int t = idx / HW;
    int c = t % Fch;
    int b = t / Fch;

    size_t base = ((size_t)b * (3 * Fch) + c) * HW + p;
    float ir = i2h[base];
    float iu = i2h[base + (size_t)Fch * HW];
    float im = i2h[base + (size_t)(2 * Fch) * HW];
    float hr = h2h[base];
    float hu = h2h[base + (size_t)Fch * HW];
    float hm = h2h[base + (size_t)(2 * Fch) * HW];

    float reset  = sigmoidf(ir + hr);
    float update = sigmoidf(iu + hu);
    float nm     = im + reset * hm;
    nm = nm > 0.0f ? nm : 0.2f * nm;
    float prev = hprev[idx];
    outT[idx] = update * prev + (1.0f - update) * nm;
}

// ---------------------------------------------------------------------------
extern "C" void kernel_launch(void* const* d_in, const int* in_sizes, int n_in,
                              void* d_out, int out_size, void* d_ws, size_t ws_size,
                              hipStream_t stream)
{
    (void)n_in; (void)out_size; (void)ws_size;

    const float* inputs  = (const float*)d_in[0];   // [S,B,8,H,W]
    const float* h0      = (const float*)d_in[1];   // [B,F,H,W]
    const float* w_i2h   = (const float*)d_in[2];   // [288,8,3,3]
    const float* b_i2h   = (const float*)d_in[3];
    const float* w_i2f   = (const float*)d_in[4];   // [32,8,5,5]
    const float* b_i2f   = (const float*)d_in[5];
    const float* w_h2f   = (const float*)d_in[6];   // [32,96,5,5]
    const float* b_h2f   = (const float*)d_in[7];
    const float* w_flows = (const float*)d_in[8];   // [10,32,5,5]
    const float* b_flows = (const float*)d_in[9];
    const float* w_ret   = (const float*)d_in[10];  // [288,480,1,1]
    const float* b_ret   = (const float*)d_in[11];

    const int S = in_sizes[0] / (Bn * CIN0 * HW);
    float* out = (float*)d_out;                     // [S][B][F][HW] + [B][F][HW]

    // ---- workspace layout --------------------------------------------------
    size_t off = 0;
    auto alloc = [&off](size_t bytes) {
        size_t r = off;
        off = (off + bytes + 255) & ~(size_t)255;
        return r;
    };
    char* ws = (char*)d_ws;
    const int N_WI2H = 288 * 8 * 9;       // 20736
    const int N_WI2F = 32 * 8 * 25;       // 6400
    const int N_WH2F = 32 * 96 * 25;      // 76800
    const int N_WFLO = 10 * 32 * 25;      // 8000
    const int N_WRET = 288 * 480;         // 138240

    __bf16* wb_i2h  = (__bf16*)(ws + alloc((size_t)N_WI2H * 2));
    __bf16* wb_i2f  = (__bf16*)(ws + alloc((size_t)N_WI2F * 2));
    __bf16* wb_h2f  = (__bf16*)(ws + alloc((size_t)N_WH2F * 2));
    __bf16* wb_flo  = (__bf16*)(ws + alloc((size_t)N_WFLO * 2));
    __bf16* wb_ret  = (__bf16*)(ws + alloc((size_t)N_WRET * 2));
    __bf16* x_bf    = (__bf16*)(ws + alloc((size_t)Bn * CIN0 * HW * 2));
    __bf16* h_bf    = (__bf16*)(ws + alloc((size_t)Bn * Fch * HW * 2));
    __bf16* f_bf    = (__bf16*)(ws + alloc((size_t)Bn * 32 * HW * 2));
    __bf16* warp_bf = (__bf16*)(ws + alloc((size_t)Bn * Lfl * Fch * HW * 2));
    float*  i2h_t   = (float*)(ws + alloc((size_t)Bn * 288 * HW * 4));
    float*  f_buf   = (float*)(ws + alloc((size_t)Bn * 32 * HW * 4));
    float*  flo_buf = (float*)(ws + alloc((size_t)Bn * 10 * HW * 4));
    float*  h2h_buf = (float*)(ws + alloc((size_t)Bn * 288 * HW * 4));

    auto cdiv = [](int a, int b) { return (a + b - 1) / b; };

    // ---- weights -> bf16, repacked [O][KH*KW][I] (once per call) -----------
    repack_w_bf16<<<cdiv(N_WI2H, 256), 256, 0, stream>>>(w_i2h, wb_i2h, 8, 9, N_WI2H);
    repack_w_bf16<<<cdiv(N_WI2F, 256), 256, 0, stream>>>(w_i2f, wb_i2f, 8, 25, N_WI2F);
    repack_w_bf16<<<cdiv(N_WH2F, 256), 256, 0, stream>>>(w_h2f, wb_h2f, 96, 25, N_WH2F);
    repack_w_bf16<<<cdiv(N_WFLO, 256), 256, 0, stream>>>(w_flows, wb_flo, 32, 25, N_WFLO);
    repack_w_bf16<<<cdiv(N_WRET, 256), 256, 0, stream>>>(w_ret, wb_ret, 480, 1, N_WRET);

    const dim3 blk(256);
    // COUT=288, MT=3 -> 48 ch/block -> grid.y = 6 (exact)
    const dim3 g288(HW / 128, 6, Bn);
    // COUT=32, MT=2 -> 32 ch/block -> grid.y = 1 (exact)
    const dim3 g32(HW / 128, 1, Bn);
    // COUT=10, MT=1 -> grid.y = 1 (guarded)
    const dim3 g10(HW / 128, 1, Bn);

    const int nX = Bn * CIN0 * HW;   // 294912
    const int nH = Bn * Fch * HW;    // 3538944
    const int nF = Bn * 32 * HW;     // 1179648

    for (int t = 0; t < S; ++t) {
        const float* x_t   = inputs + (size_t)t * nX;
        const float* hprev = (t == 0) ? h0 : (out + (size_t)(t - 1) * BFHW);

        nchw_to_nhwc_bf16<<<cdiv(nX, 256), 256, 0, stream>>>(x_t, x_bf, CIN0, nX);
        nchw_to_nhwc_bf16<<<cdiv(nH, 256), 256, 0, stream>>>(hprev, h_bf, Fch, nH);

        // i2h: 3x3, 8->288
        conv_wmma<8, 3, 3, 1, 3><<<g288, blk, 0, stream>>>(
            x_bf, wb_i2h, b_i2h, i2h_t, 288, 0, 0);
        // f = leaky(i2f(x) + h2f(h))
        conv_wmma<8, 5, 5, 2, 2><<<g32, blk, 0, stream>>>(
            x_bf, wb_i2f, b_i2f, f_buf, 32, 0, 0);
        conv_wmma<96, 5, 5, 2, 2><<<g32, blk, 0, stream>>>(
            h_bf, wb_h2f, b_h2f, f_buf, 32, 1, 1);
        nchw_to_nhwc_bf16<<<cdiv(nF, 256), 256, 0, stream>>>(f_buf, f_bf, 32, nF);
        // flows: 5x5, 32->10
        conv_wmma<32, 5, 5, 2, 1><<<g10, blk, 0, stream>>>(
            f_bf, wb_flo, b_flows, flo_buf, 10, 0, 0);
        // L flow-warped copies of prev_h -> NHWC [B][HW][480]
        warp_kernel<<<cdiv(Bn * Lfl * HW, 256), 256, 0, stream>>>(
            hprev, flo_buf, warp_bf);
        // ret: 1x1, 480->288 (B fragments are pure v16bf loads)
        conv_wmma<480, 1, 1, 0, 3><<<g288, blk, 0, stream>>>(
            warp_bf, wb_ret, b_ret, h2h_buf, 288, 0, 0);
        // fused GRU gates -> outputs[t] (also prev_h for t+1)
        gate_kernel<<<cdiv(BFHW, 256), 256, 0, stream>>>(
            i2h_t, h2h_buf, hprev, out + (size_t)t * BFHW);
    }

    // final_h = outputs[S-1]
    copy_f32<<<cdiv(BFHW, 256), 256, 0, stream>>>(
        out + (size_t)(S - 1) * BFHW, out + (size_t)S * BFHW, BFHW);
}
